// MultiHeadAttention_18717467476233
// MI455X (gfx1250) — matrix operand order repaired and checked by reading, not compiled, not run on hardware
//
#include <hip/hip_runtime.h>

// Problem constants (match reference)
#define BB   4
#define SS   1024
#define DD   1024
#define HH   16
#define DKK  64

#define USE_TDM 1   // stage GEMM tiles with tensor_load_to_lds (TDM) if 1

typedef __attribute__((ext_vector_type(16))) _Float16 v16h;
typedef __attribute__((ext_vector_type(8)))  _Float16 v8h;
typedef __attribute__((ext_vector_type(4)))  _Float16 v4h;
typedef __attribute__((ext_vector_type(8)))  float    v8f;
typedef __attribute__((ext_vector_type(4)))  unsigned int u32x4;
typedef __attribute__((ext_vector_type(8)))  int      i32x8;
typedef __attribute__((ext_vector_type(4)))  int      i32x4;

#define MTILE 128
#define KSTEP 32
#define LDSS  40   // padded LDS row stride in halves (80B) == TDM pad 16DW+4DW

// ---------------------------------------------------------------------------
// TDM: DMA a (rows x KSTEP) f16 tile (row stride ld elements) into LDS at
// byte address ldsAddr, padding each 64B row with 16B -> LDSS stride.
// D# packing per CDNA5 ISA ch.8 (group0: count/lds/global/type, group1: dims).
// This toolchain's builtin takes 6 args (g0, g1, g2, g3, g4, cpol).
// ---------------------------------------------------------------------------
__device__ __forceinline__ void tdm_load_tile_f16(const _Float16* g, int ld,
                                                  unsigned ldsAddr, int rows)
{
#if USE_TDM
    const unsigned long long ga = (unsigned long long)(uintptr_t)g;
    u32x4 g0;
    g0[0] = 1u;                                        // count=1, user mode
    g0[1] = ldsAddr;                                   // LDS byte address
    g0[2] = (unsigned)(ga & 0xffffffffu);              // global_addr[31:0]
    g0[3] = (unsigned)((ga >> 32) & 0x01ffffffu)       // global_addr[56:32]
          | (2u << 30);                                // type=2 (image)
    i32x8 g1;
    g1[0] = (1 << 16)                                  // data_size=1 (2 bytes)
          | (1 << 20)                                  // pad_enable
          | (3 << 22)                                  // pad_interval: 16 DW
          | (3 << 25);                                 // pad_amount: 4 DW
    g1[1] = (KSTEP & 0xffff) << 16;                    // tensor_dim0 lo
    g1[2] = (rows  & 0xffff) << 16;                    // dim0 hi=0 | tensor_dim1 lo
    g1[3] = (KSTEP & 0xffff) << 16;                    // dim1 hi=0 | tile_dim0
    g1[4] = (rows  & 0xffff);                          // tile_dim1 | tile_dim2=0
    g1[5] = ld;                                        // tensor_dim0_stride lo
    g1[6] = 0;                                         // stride hi | dim1_stride lo
    g1[7] = 0;
    const i32x4 z4 = {0, 0, 0, 0};                     // groups 2/3 unused (2D)
    const i32x8 z8 = {0, 0, 0, 0, 0, 0, 0, 0};
    __builtin_amdgcn_tensor_load_to_lds(g0, g1, z4, z4, z8, 0);
#endif
}

// ---------------------------------------------------------------------------
// Batched GEMM: C = alpha * A(MxK,f16) * B(NxK,f16)^T [+ R(f32)], TC = f32|f16.
// B is always stored N x K row-major (K contiguous). Block tile MTILE x BN,
// 256 threads = 8 waves as (8/(BN/64)) x (BN/64); wave owns (MTILE/WY) x 64.
// Tiles staged to double-buffered LDS by TDM (wave 0), or manual fallback.
// transC: store C[col*ldc + row] instead of C[row*ldc + col].
// ---------------------------------------------------------------------------
template <int BN, typename TC>
__global__ __launch_bounds__(256)
void gemm_wmma_f16(const _Float16* __restrict__ A, const _Float16* __restrict__ Bm,
                   TC* __restrict__ C, const float* __restrict__ R,
                   int K, int lda, int ldb, int ldc,
                   float alpha, int transC, int nH,
                   long long aSB, long long aSH,
                   long long bSB, long long bSH,
                   long long cSB, long long cSH)
{
    constexpr int WX     = BN / 64;
    constexpr int WY     = 8 / WX;
    constexpr int AFRAGS = MTILE / (WY * 16);
    constexpr int ASZ    = MTILE * LDSS;   // halves
    constexpr int BSZ    = BN * LDSS;

    __shared__ __align__(16) _Float16 As[2 * ASZ];
    __shared__ __align__(16) _Float16 Bs[2 * BSZ];

    const int bh = blockIdx.z;
    const int bb = bh / nH;
    const int hh = bh % nH;
    A  += (long long)bb * aSB + (long long)hh * aSH;
    Bm += (long long)bb * bSB + (long long)hh * bSH;
    C  += (long long)bb * cSB + (long long)hh * cSH;

    const int tid   = threadIdx.x;
    const int lane  = tid & 31;
    const int wave  = tid >> 5;
    const int m0    = blockIdx.y * MTILE;
    const int n0    = blockIdx.x * BN;
    const int wy    = wave / WX;
    const int wx    = wave % WX;
    const int waveM = wy * (AFRAGS * 16);
    const int waveN = wx * 64;

    const _Float16* aT = A + (long long)m0 * lda;   // tile origin, k advances
    const _Float16* bT = Bm + (long long)n0 * ldb;
    const unsigned asAddr = (unsigned)(uintptr_t)(const void*)&As[0];
    const unsigned bsAddr = (unsigned)(uintptr_t)(const void*)&Bs[0];

    v8f zero = {};
    v8f acc[AFRAGS][4];
#pragma unroll
    for (int s = 0; s < AFRAGS; ++s)
#pragma unroll
        for (int nt = 0; nt < 4; ++nt) acc[s][nt] = zero;

#if USE_TDM
    if (wave == 0) {
        tdm_load_tile_f16(aT, lda, asAddr, MTILE);
        tdm_load_tile_f16(bT, ldb, bsAddr, BN);
    }
#endif

    int buf = 0;
    for (int k0 = 0; k0 < K; k0 += KSTEP) {
        const _Float16* as = &As[buf * ASZ];
        const _Float16* bs = &Bs[buf * BSZ];
#if USE_TDM
        if (wave == 0) __builtin_amdgcn_s_wait_tensorcnt(0);
        __syncthreads();
        const int kn = k0 + KSTEP;
        if (kn < K && wave == 0) {
            tdm_load_tile_f16(aT + kn, lda, asAddr + (unsigned)((buf ^ 1) * ASZ * 2), MTILE);
            tdm_load_tile_f16(bT + kn, ldb, bsAddr + (unsigned)((buf ^ 1) * BSZ * 2), BN);
        }
#else
        __syncthreads();
        {   // manual f16 staging fallback (single-buffered semantics)
            _Float16* aw = &As[buf * ASZ];
            for (int c = tid; c < MTILE * 2; c += 256) {
                const int r = c >> 1, o = (c & 1) * 16;
                const _Float16* src = aT + (long long)r * lda + k0 + o;
                *(v8h*)&aw[r * LDSS + o]     = *(const v8h*)(src);
                *(v8h*)&aw[r * LDSS + o + 8] = *(const v8h*)(src + 8);
            }
            _Float16* bw = &Bs[buf * BSZ];
            for (int c = tid; c < BN * 2; c += 256) {
                const int r = c >> 1, o = (c & 1) * 16;
                const _Float16* src = bT + (long long)r * ldb + k0 + o;
                *(v8h*)&bw[r * LDSS + o]     = *(const v8h*)(src);
                *(v8h*)&bw[r * LDSS + o + 8] = *(const v8h*)(src + 8);
            }
        }
        __syncthreads();
#endif
        // ---- compute from LDS(buf) --------------------------------------
        // A frag (16x32): lane<16 -> K 0-7 & 16-23 ; lane>=16 -> 8-15 & 24-31
        v16h af[AFRAGS];
#pragma unroll
        for (int s = 0; s < AFRAGS; ++s) {
            const int arow  = waveM + s * 16 + (lane & 15);
            const int akoff = (lane < 16) ? 0 : 8;
            const _Float16* ab = &as[arow * LDSS + akoff];
            const v8h a0 = *(const v8h*)(ab);
            const v8h a1 = *(const v8h*)(ab + 16);
#pragma unroll
            for (int i = 0; i < 8; ++i) { af[s][i] = a0[i]; af[s][i + 8] = a1[i]; }
        }
        // B frag (32x16): lane<16 -> K 0-7 & 16-23 ; lane>=16 -> 8-15 & 24-31
        const int bkoff = (lane < 16) ? 0 : 8;
#pragma unroll
        for (int nt = 0; nt < 4; ++nt) {
            const _Float16* bbase = &bs[(waveN + nt * 16 + (lane & 15)) * LDSS + bkoff];
            const v8h b0 = *(const v8h*)(bbase);
            const v8h b1 = *(const v8h*)(bbase + 16);
            v16h bf;
#pragma unroll
            for (int i = 0; i < 8; ++i) { bf[i] = b0[i]; bf[i + 8] = b1[i]; }
#pragma unroll
            for (int s = 0; s < AFRAGS; ++s)
                acc[s][nt] = __builtin_amdgcn_wmma_f32_16x16x32_f16(
                    false, af[s], false, bf, (short)0, acc[s][nt], false, false);
        }
        buf ^= 1;
    }

    // ---- epilogue: C layout VGPR r -> M = r (lanes 0-15) / r+8 (16-31) ---
#pragma unroll
    for (int s = 0; s < AFRAGS; ++s) {
#pragma unroll
        for (int nt = 0; nt < 4; ++nt) {
            const int col = n0 + waveN + nt * 16 + (lane & 15);
#pragma unroll
            for (int r = 0; r < 8; ++r) {
                const int row = m0 + waveM + s * 16 + ((lane < 16) ? r : r + 8);
                const long long off = transC ? (long long)col * ldc + row
                                             : (long long)row * ldc + col;
                float vv = acc[s][nt][r] * alpha;
                if (R) vv += R[off];
                C[off] = (TC)vv;
            }
        }
    }
}

// ---------------------------------------------------------------------------
// Elementwise f32 -> f16 (n multiple of 1024)
// ---------------------------------------------------------------------------
__global__ __launch_bounds__(256)
void cvt16(const float* __restrict__ in, _Float16* __restrict__ out)
{
    const long long i = ((long long)blockIdx.x * 256 + threadIdx.x) * 4;
    const float4 f = *(const float4*)&in[i];
    v4h o;
    o[0] = (_Float16)f.x; o[1] = (_Float16)f.y;
    o[2] = (_Float16)f.z; o[3] = (_Float16)f.w;
    *(v4h*)&out[i] = o;
}

// ---------------------------------------------------------------------------
// Transposing f32 -> f16: out[n*rows + k] = in[k*cols + n]. 32x32 LDS tiles.
// ---------------------------------------------------------------------------
__global__ __launch_bounds__(256)
void cvtT16(const float* __restrict__ in, _Float16* __restrict__ out,
            int rows, int cols)
{
    __shared__ _Float16 t[32][33];
    const int bn = blockIdx.x * 32;   // n (output row)
    const int bk = blockIdx.y * 32;   // k (output col)
    const int tx = threadIdx.x, ty = threadIdx.y;   // (32, 8)
#pragma unroll
    for (int j = 0; j < 4; ++j)
        t[ty + j * 8][tx] = (_Float16)in[(long long)(bk + ty + j * 8) * cols + bn + tx];
    __syncthreads();
#pragma unroll
    for (int j = 0; j < 4; ++j)
        out[(long long)(bn + ty + j * 8) * rows + bk + tx] = t[tx][ty + j * 8];
}

// ---------------------------------------------------------------------------
// Masked softmax over last axis, in place (f32) + f16 copy for attn@V.
// ---------------------------------------------------------------------------
__global__ __launch_bounds__(256)
void softmax_mask(float* __restrict__ attn, _Float16* __restrict__ attn16,
                  const int* __restrict__ mask)
{
    __shared__ float red[8];
    const long long row = blockIdx.x;                 // 0 .. B*H*S-1
    const int q = (int)(row % SS);
    const int b = (int)(row / ((long long)HH * SS));
    float* p = attn + row * SS;
    _Float16* p16 = attn16 + row * SS;
    const int* m = mask + ((long long)b * SS + q) * SS;
    const int tid = threadIdx.x;

    float v[4];
    float mx = -3.402823e38f;
#pragma unroll
    for (int j = 0; j < 4; ++j) {
        const int idx = tid + j * 256;
        const float x = (m[idx] == 0) ? -1.0e9f : p[idx];
        v[j] = x;
        mx = fmaxf(mx, x);
    }
#pragma unroll
    for (int off = 16; off > 0; off >>= 1) mx = fmaxf(mx, __shfl_xor(mx, off, 32));
    if ((tid & 31) == 0) red[tid >> 5] = mx;
    __syncthreads();
    float bmx = red[0];
#pragma unroll
    for (int i = 1; i < 8; ++i) bmx = fmaxf(bmx, red[i]);
    __syncthreads();

    float s = 0.f;
#pragma unroll
    for (int j = 0; j < 4; ++j) { v[j] = __expf(v[j] - bmx); s += v[j]; }
#pragma unroll
    for (int off = 16; off > 0; off >>= 1) s += __shfl_xor(s, off, 32);
    if ((tid & 31) == 0) red[tid >> 5] = s;
    __syncthreads();
    float bs = 0.f;
#pragma unroll
    for (int i = 0; i < 8; ++i) bs += red[i];
    const float inv = 1.0f / bs;
#pragma unroll
    for (int j = 0; j < 4; ++j) {
        const int idx = tid + j * 256;
        const float pv = v[j] * inv;
        p[idx] = pv;
        p16[idx] = (_Float16)pv;
    }
}

// ---------------------------------------------------------------------------
// In-place LayerNorm over D, one block per row. Biased variance, eps=1e-6.
// ---------------------------------------------------------------------------
__global__ __launch_bounds__(256)
void layernorm_inplace(float* __restrict__ out,
                       const float* __restrict__ gamma,
                       const float* __restrict__ beta)
{
    __shared__ float r1[8], r2[8];
    float* x = out + (long long)blockIdx.x * DD;
    const int tid = threadIdx.x;

    float v[4];
    float s = 0.f, s2 = 0.f;
#pragma unroll
    for (int j = 0; j < 4; ++j) {
        v[j] = x[tid + j * 256];
        s  += v[j];
        s2 += v[j] * v[j];
    }
#pragma unroll
    for (int off = 16; off > 0; off >>= 1) { s += __shfl_xor(s, off, 32); s2 += __shfl_xor(s2, off, 32); }
    if ((tid & 31) == 0) { r1[tid >> 5] = s; r2[tid >> 5] = s2; }
    __syncthreads();
    float ts = 0.f, ts2 = 0.f;
#pragma unroll
    for (int i = 0; i < 8; ++i) { ts += r1[i]; ts2 += r2[i]; }
    const float mean = ts * (1.0f / DD);
    const float var  = ts2 * (1.0f / DD) - mean * mean;
    const float rstd = rsqrtf(var + 1e-6f);
#pragma unroll
    for (int j = 0; j < 4; ++j) {
        const int idx = tid + j * 256;
        x[idx] = (v[j] - mean) * rstd * gamma[idx] + beta[idx];
    }
}

// ---------------------------------------------------------------------------
extern "C" void kernel_launch(void* const* d_in, const int* in_sizes, int n_in,
                              void* d_out, int out_size, void* d_ws, size_t ws_size,
                              hipStream_t stream)
{
    const float* q     = (const float*)d_in[0];
    const float* k     = (const float*)d_in[1];
    const float* v     = (const float*)d_in[2];
    const int*   mask  = (const int*)  d_in[3];
    const float* w_qs  = (const float*)d_in[4];
    const float* w_ks  = (const float*)d_in[5];
    const float* w_vs  = (const float*)d_in[6];
    const float* w_fc  = (const float*)d_in[7];
    const float* gamma = (const float*)d_in[8];
    const float* beta  = (const float*)d_in[9];

    float* outp = (float*)d_out;                              // (B,S,D)
    float* attn = outp + (long long)BB * SS * DD;             // (B,H,S,S)

    // ---- f16 workspace layout -------------------------------------------
    const long long NQ = (long long)BB * SS * DD;             // 4M elements
    const long long NW = (long long)DD * DD;                  // 1M elements
    _Float16* q16    = (_Float16*)d_ws;
    _Float16* k16    = q16  + NQ;
    _Float16* v16    = k16  + NQ;
    _Float16* wqT    = v16  + NQ;
    _Float16* wkT    = wqT  + NW;
    _Float16* wvT    = wkT  + NW;
    _Float16* wfT    = wvT  + NW;
    _Float16* qh16   = wfT  + NW;
    _Float16* kh16   = qh16 + NQ;
    _Float16* vhT16  = kh16 + NQ;   // (B, H*DV, S): K-contiguous for attn@V
    _Float16* ctx16  = vhT16 + NQ;
    _Float16* attn16 = ctx16 + NQ;  // (B,H,S,S) f16 probs

    const dim3 blk(256);
    const int Mrows = BB * SS;                                 // 4096

    // 0) one-time f32->f16 conversions (+ weight transposes to N x K)
    cvt16<<<dim3((unsigned)(NQ / 1024)), blk, 0, stream>>>(q, q16);
    cvt16<<<dim3((unsigned)(NQ / 1024)), blk, 0, stream>>>(k, k16);
    cvt16<<<dim3((unsigned)(NQ / 1024)), blk, 0, stream>>>(v, v16);
    const dim3 tblk(32, 8), tgrd(DD / 32, DD / 32);
    cvtT16<<<tgrd, tblk, 0, stream>>>(w_qs, wqT, DD, DD);
    cvtT16<<<tgrd, tblk, 0, stream>>>(w_ks, wkT, DD, DD);
    cvtT16<<<tgrd, tblk, 0, stream>>>(w_vs, wvT, DD, DD);
    cvtT16<<<tgrd, tblk, 0, stream>>>(w_fc, wfT, DD, DD);

    // 1-2) Q/K projections -> (B,S,H*dk) f16
    const dim3 gproj(DD / 128, Mrows / MTILE, 1);
    gemm_wmma_f16<128, _Float16><<<gproj, blk, 0, stream>>>(q16, wqT, qh16, nullptr,
        DD, DD, DD, DD, 1.0f, 0, 1, 0, 0, 0, 0, 0, 0);
    gemm_wmma_f16<128, _Float16><<<gproj, blk, 0, stream>>>(k16, wkT, kh16, nullptr,
        DD, DD, DD, DD, 1.0f, 0, 1, 0, 0, 0, 0, 0, 0);

    // 3) V projection, written transposed per batch: vhT (B, D, S)
    const dim3 gvproj(DD / 128, SS / MTILE, BB);
    gemm_wmma_f16<128, _Float16><<<gvproj, blk, 0, stream>>>(v16, wvT, vhT16, nullptr,
        DD, DD, DD, SS, 1.0f, 1, 1,
        (long long)SS * DD, 0, 0, 0, (long long)DD * SS, 0);

    // 4) scores = Qh @ Kh^T / sqrt(dk), batched over B*H -> attn (f32 out)
    const dim3 gsc(SS / 128, SS / MTILE, BB * HH);
    gemm_wmma_f16<128, float><<<gsc, blk, 0, stream>>>(qh16, kh16, attn, nullptr,
        DKK, DD, DD, SS, 0.125f, 0, HH,
        (long long)SS * DD, (long long)DKK,
        (long long)SS * DD, (long long)DKK,
        (long long)HH * SS * SS, (long long)SS * SS);

    // 5) masked softmax in place (f32) + f16 copy
    softmax_mask<<<dim3(BB * HH * SS), blk, 0, stream>>>(attn, attn16, mask);

    // 6) ctx = attn @ Vh (B = vhT, K-contiguous), N=64 tile, f16 out
    const dim3 gav(1, SS / MTILE, BB * HH);
    gemm_wmma_f16<64, _Float16><<<gav, blk, 0, stream>>>(attn16, vhT16, ctx16, nullptr,
        SS, SS, SS, DD, 1.0f, 0, HH,
        (long long)HH * SS * SS, (long long)SS * SS,
        (long long)DD * SS, (long long)DKK * SS,
        (long long)SS * DD, (long long)DKK);

    // 7) out = ctx @ w_fc + residual(q), f32 out
    const dim3 gfc(DD / 128, Mrows / MTILE, 1);
    gemm_wmma_f16<128, float><<<gfc, blk, 0, stream>>>(ctx16, wfT, outp, q,
        DD, DD, DD, DD, 1.0f, 0, 1, 0, 0, 0, 0, 0, 0);

    // 8) LayerNorm in place
    layernorm_inplace<<<dim3(BB * SS), blk, 0, stream>>>(outp, gamma, beta);
}
